// GAEncoder_21122649162479
// MI455X (gfx1250) — matrix operand order repaired and checked by reading, not compiled, not run on hardware
//
#include <hip/hip_runtime.h>
#include <hip/hip_bf16.h>

// ---------------------------------------------------------------------------
// 2-layer GCN on gfx1250 (MI455X).  Memory-bound on edge aggregation; GEMMs
// run through v_wmma_f32_16x16x32_bf16 so they are negligible.  Edge scatter
// uses inline-asm non-returning global_atomic_add_f32 (guaranteed native RMW,
// no CAS loop fallback).
// ---------------------------------------------------------------------------

typedef __attribute__((ext_vector_type(16))) __bf16 v16bf;
typedef __attribute__((ext_vector_type(8)))  __bf16 v8bf;
typedef __attribute__((ext_vector_type(8)))  float  v8f;

static constexpr int IN_F  = 256;
static constexpr int H_F   = 128;
static constexpr int OUT_F = 64;

__device__ __forceinline__ unsigned short f2bf(float f) {
  union { float f; unsigned u; } v; v.f = f;
  unsigned u = v.u;
  u += 0x7FFFu + ((u >> 16) & 1u);   // round-to-nearest-even
  return (unsigned short)(u >> 16);
}

// Non-returning float atomic add: single GLOBAL_ATOMIC_ADD_F32 (STOREcnt
// tracked; s_endpgm's implicit wait-idle covers completion).
__device__ __forceinline__ void atomic_add_f32(float* p, float v) {
  asm volatile("global_atomic_add_f32 %0, %1, off"
               :
               : "v"(p), "v"(v)
               : "memory");
}

// ------------------------------ utility kernels -----------------------------

__global__ void fill_f32(float* __restrict__ p, float v, long n) {
  long i = (long)blockIdx.x * blockDim.x + threadIdx.x;
  if (i < n) p[i] = v;
}

__global__ void deg_kernel(const long long* __restrict__ dst,
                           float* __restrict__ deg, int E) {
  int e = blockIdx.x * blockDim.x + threadIdx.x;
  if (e < E) atomic_add_f32(&deg[(int)dst[e]], 1.0f);
}

__global__ void rsqrt_kernel(float* __restrict__ d, int n) {
  int i = blockIdx.x * blockDim.x + threadIdx.x;
  if (i < n) d[i] = rsqrtf(d[i]);           // deg >= 1 always (self-loop)
}

// Transposed bf16 weights: W1T[h][k] (128x256), W2T[o][k] (64x128).
__global__ void prep_weights(const float* __restrict__ W1,
                             const float* __restrict__ W2,
                             unsigned short* __restrict__ W1T,
                             unsigned short* __restrict__ W2T) {
  int i = blockIdx.x * blockDim.x + threadIdx.x;
  if (i < IN_F * H_F) {
    int k = i / H_F, h = i % H_F;
    W1T[h * IN_F + k] = f2bf(W1[i]);
  } else {
    int j = i - IN_F * H_F;
    if (j < H_F * OUT_F) {
      int k = j / OUT_F, o = j % OUT_F;
      W2T[o * H_F + k] = f2bf(W2[j]);
    }
  }
}

__global__ void conv_bf16(const float* __restrict__ x,
                          unsigned short* __restrict__ xb, long n) {
  long i = (long)blockIdx.x * blockDim.x + threadIdx.x;
  if (i < n) xb[i] = f2bf(x[i]);
}

// ------------------------------- WMMA GEMM ----------------------------------
// C[M x NC] = A[M x K] * B[K x NC], A row-major bf16, B provided transposed
// (BT[NC x K] bf16, row-major => a B column is contiguous).  One wave per
// 16x16 output tile; blockDim = 32 * (NC/16); grid.x = M/16 (M % 16 == 0).
// Fragment layouts per CDNA5 ISA 7.12.2 (16-bit A 16x32, B 32x16, f32 C/D).

template <int K, int NC>
__global__ void gemm_bf16_wmma(const unsigned short* __restrict__ A,
                               const unsigned short* __restrict__ BT,
                               float* __restrict__ C) {
  const int lane = threadIdx.x & 31;
  const int wave = threadIdx.x >> 5;
  const int m0   = blockIdx.x * 16;
  const int n0   = wave * 16;
  const int sel  = lane >> 4;     // half-wave select
  const int l15  = lane & 15;

  const unsigned short* arow = A  + (long)(m0 + l15) * K;   // A row (M = lane)
  const unsigned short* bcol = BT + (long)(n0 + l15) * K;   // B column (N = lane)

  v8f acc = {};
#pragma unroll
  for (int kb = 0; kb < K; kb += 32) {
    // A 16x32 bf16: lanes 0-15 hold K {kb..kb+7, kb+16..kb+23} of row M,
    // lanes 16-31 hold K {kb+8..kb+15, kb+24..kb+31}.
    v8bf alo = *(const v8bf*)(arow + kb + sel * 8);
    v8bf ahi = *(const v8bf*)(arow + kb + 16 + sel * 8);
    // B 32x16 bf16: lanes 0-15 hold K {kb..kb+7, kb+16..kb+23} of col N,
    // lanes 16-31 hold K {kb+8..kb+15, kb+24..kb+31}.
    v8bf blo = *(const v8bf*)(bcol + kb + sel * 8);
    v8bf bhi = *(const v8bf*)(bcol + kb + 16 + sel * 8);
    v16bf a = __builtin_shufflevector(alo, ahi, 0,1,2,3,4,5,6,7,8,9,10,11,12,13,14,15);
    v16bf b = __builtin_shufflevector(blo, bhi, 0,1,2,3,4,5,6,7,8,9,10,11,12,13,14,15);
    acc = __builtin_amdgcn_wmma_f32_16x16x32_bf16(
        /*neg_a=*/false, a, /*neg_b=*/false, b,
        /*c_mod=*/(short)0, acc, /*reuse_a=*/false, /*reuse_b=*/false);
  }

  // D: VGPR r holds M = r (lanes 0-15) / r+8 (lanes 16-31), N = lane&15.
  float* crow = C + (long)(m0 + sel * 8) * NC + n0 + l15;
#pragma unroll
  for (int r = 0; r < 8; ++r)
    crow[(long)r * NC] = acc[r];
}

// ----------------------------- edge aggregation -----------------------------
// One wave per edge; FPL = F/32 features per lane.  agg[dst] += h[src] * norm.

template <int F>
__global__ void edge_agg(const long long* __restrict__ src,
                         const long long* __restrict__ dst,
                         const float* __restrict__ dinv,
                         const float* __restrict__ h,
                         float* __restrict__ agg, int E) {
  long gid = (long)blockIdx.x * blockDim.x + threadIdx.x;
  int e    = (int)(gid >> 5);
  int lane = (int)(gid & 31);
  if (e >= E) return;
  int s = (int)src[e];
  int d = (int)dst[e];
  float nm = dinv[s] * dinv[d];
  constexpr int FPL = F / 32;
  const float* hp = h   + (long)s * F + lane * FPL;
  float*       ap = agg + (long)d * F + lane * FPL;
  if constexpr (FPL == 4) {
    float4 v = *(const float4*)hp;
    atomic_add_f32(ap + 0, v.x * nm);
    atomic_add_f32(ap + 1, v.y * nm);
    atomic_add_f32(ap + 2, v.z * nm);
    atomic_add_f32(ap + 3, v.w * nm);
  } else {
    float2 v = *(const float2*)hp;
    atomic_add_f32(ap + 0, v.x * nm);
    atomic_add_f32(ap + 1, v.y * nm);
  }
}

// ------------------------------- epilogues ----------------------------------
// Layer 1: self-loop term + bias + ReLU, emit bf16 activation for GEMM2.
__global__ void postact_relu_bf16(const float* __restrict__ agg,
                                  const float* __restrict__ h,
                                  const float* __restrict__ dinv,
                                  const float* __restrict__ b,
                                  unsigned short* __restrict__ outbf, long n) {
  long i = (long)blockIdx.x * blockDim.x + threadIdx.x;
  if (i >= n) return;
  int node = (int)(i / H_F), f = (int)(i % H_F);
  float di = dinv[node];
  float v  = agg[i] + h[i] * di * di + b[f];
  outbf[i] = f2bf(fmaxf(v, 0.0f));
}

// Layer 2: self-loop term + bias, fp32 output.
__global__ void final_kernel(const float* __restrict__ agg,
                             const float* __restrict__ g,
                             const float* __restrict__ dinv,
                             const float* __restrict__ b,
                             float* __restrict__ out, long n) {
  long i = (long)blockIdx.x * blockDim.x + threadIdx.x;
  if (i >= n) return;
  int node = (int)(i / OUT_F), f = (int)(i % OUT_F);
  float di = dinv[node];
  out[i] = agg[i] + g[i] * di * di + b[f];
}

// ------------------------------- launcher -----------------------------------

extern "C" void kernel_launch(void* const* d_in, const int* in_sizes, int n_in,
                              void* d_out, int out_size, void* d_ws, size_t ws_size,
                              hipStream_t stream) {
  const float*     x  = (const float*)d_in[0];
  const long long* ei = (const long long*)d_in[1];   // int64 [2, E]
  const float*     W1 = (const float*)d_in[2];
  const float*     b1 = (const float*)d_in[3];
  const float*     W2 = (const float*)d_in[4];
  const float*     b2 = (const float*)d_in[5];

  const int N = in_sizes[0] / IN_F;   // 50000 (divisible by 16)
  const int E = in_sizes[1] / 2;      // 800000
  const long long* srcp = ei;
  const long long* dstp = ei + E;

  char* ws = (char*)d_ws;
  auto al = [](size_t v) { return (v + 255) & ~(size_t)255; };
  size_t off = 0;
  float*          dinv = (float*)(ws + off);          off = al(off + (size_t)N * 4);
  unsigned short* W1T  = (unsigned short*)(ws + off); off = al(off + (size_t)IN_F * H_F * 2);
  unsigned short* W2T  = (unsigned short*)(ws + off); off = al(off + (size_t)H_F * OUT_F * 2);
  unsigned short* bufA = (unsigned short*)(ws + off); off = al(off + (size_t)N * IN_F * 2);
  float*          bufB = (float*)(ws + off);          off = al(off + (size_t)N * H_F * 4);
  float*          bufC = (float*)(ws + off);          off = al(off + (size_t)N * H_F * 4);

  // buffer overlays (stream-ordered lifetimes)
  unsigned short* xbf    = bufA;   // phase 2 only
  unsigned short* h1relu = bufA;   // written phase 4, read phase 5
  float* h1   = bufB;              // phases 2-4
  float* g    = bufB;              // phases 5-7
  float* agg1 = bufC;              // phases 3-4
  float* agg2 = bufC;              // phases 6-7 (re-zeroed after phase 4)

  const int T = 256;
  auto blk = [&](long n) { return (int)((n + T - 1) / T); };

  // 1. degree (init 1.0 for self-loop) + zero agg1
  fill_f32<<<blk(N), T, 0, stream>>>(dinv, 1.0f, N);
  fill_f32<<<blk((long)N * H_F), T, 0, stream>>>(agg1, 0.0f, (long)N * H_F);
  deg_kernel<<<blk(E), T, 0, stream>>>(dstp, dinv, E);
  rsqrt_kernel<<<blk(N), T, 0, stream>>>(dinv, N);

  // 2. weight/activation prep (bf16), then GEMM1: h1 = x @ W1
  prep_weights<<<blk(IN_F * H_F + H_F * OUT_F), T, 0, stream>>>(W1, W2, W1T, W2T);
  conv_bf16<<<blk((long)N * IN_F), T, 0, stream>>>(x, xbf, (long)N * IN_F);
  gemm_bf16_wmma<IN_F, H_F><<<N / 16, 32 * (H_F / 16), 0, stream>>>(xbf, W1T, h1);

  // 3. edge aggregation layer 1 (1 wave / edge, 4 feats / lane)
  edge_agg<H_F><<<blk((long)E * 32), T, 0, stream>>>(srcp, dstp, dinv, h1, agg1, E);

  // 4. self-loop + bias + ReLU -> bf16 activation
  postact_relu_bf16<<<blk((long)N * H_F), T, 0, stream>>>(agg1, h1, dinv, b1,
                                                          h1relu, (long)N * H_F);

  // 5. zero agg2, GEMM2: g = h1relu @ W2
  fill_f32<<<blk((long)N * OUT_F), T, 0, stream>>>(agg2, 0.0f, (long)N * OUT_F);
  gemm_bf16_wmma<H_F, OUT_F><<<N / 16, 32 * (OUT_F / 16), 0, stream>>>(h1relu, W2T, g);

  // 6. edge aggregation layer 2 (1 wave / edge, 2 feats / lane)
  edge_agg<OUT_F><<<blk((long)E * 32), T, 0, stream>>>(srcp, dstp, dinv, g, agg2, E);

  // 7. self-loop + bias -> output
  final_kernel<<<blk((long)N * OUT_F), T, 0, stream>>>(agg2, g, dinv, b2,
                                                       (float*)d_out, (long)N * OUT_F);
}
